// Attention_Critic_60911226192482
// MI455X (gfx1250) — matrix operand order repaired and checked by reading, not compiled, not run on hardware
//
#include <hip/hip_runtime.h>
#include <math.h>

// ---- problem constants (from the reference) -------------------------------
#define SD      96          // S_DIM
#define AD      32          // A_DIM
#define ATT     128         // ATT_DIM
#define NOTH    7           // N_AGENTS - 1
#define SROW    768         // N_AGENTS * S_DIM
#define AROW    256         // N_AGENTS * A_DIM
#define ROWS_PER_WAVE 16
#define WAVES_PER_BLK 4
#define NTHREADS      (WAVES_PER_BLK * 32)
#define ROWS_PER_BLK  (ROWS_PER_WAVE * WAVES_PER_BLK)

typedef __attribute__((ext_vector_type(16))) __bf16 v16bf;
typedef __attribute__((ext_vector_type(8)))  __bf16 v8bf;
typedef __attribute__((ext_vector_type(8)))  float  v8f;

// Intra-wave LDS producer->consumer fence (CDNA5 split counters). Same-wave
// LDS ops are in-order and lanes are lockstep, so a dscnt wait suffices.
#define WAVE_LDS_FENCE() asm volatile("s_wait_dscnt 0" ::: "memory")

__device__ __forceinline__ float leakyf(float x) { return x > 0.f ? x : 0.01f * x; }

// ---- weight staging: global f32 row-major -> LDS bf16 pre-swizzled frags --
// B-fragment layout (ISA 7.12.2, 32x16 16-bit B): lane L -> col N = L%16,
// element e<8 -> K = 8*(L/16) + e, e>=8 -> K = 16 + 8*(L/16) + (e-8).  Fragment f = t*KC + kc stored as
// 32 lanes x 16 contiguous bf16, so a fragment read is one 32-byte DS load.
template<int K, int N>
__device__ __forceinline__ void stage_weight(__bf16* dst, const float* __restrict__ W, int tid) {
  constexpr int KC = K / 32;
#pragma unroll 4
  for (int idx = tid; idx < K * N; idx += NTHREADS) {   // coalesced over n
    const int k = idx / N, n = idx % N;
    const int r = k & 31;
    const int f = (n >> 4) * KC + (k >> 5);
    const int lane = (((r >> 3) & 1) << 4) | (n & 15);
    dst[((f * 32 + lane) << 4) + ((r & 7) | ((r >> 4) << 3))] = (__bf16)W[idx];
  }
}

#define STAGE(dst, W, K, N)                                  \
  do {                                                       \
    __syncthreads();                                         \
    stage_weight<K, N>((dst), (W), threadIdx.x);             \
    __syncthreads();                                         \
  } while (0)

// ---- A-fragment loaders (ISA 7.12.2 16-bit A layout, wave32) --------------
// lane L -> row M = L%16; element e<8 -> K = k0 + 8*(L/16) + e,
//                         element e>=8 -> K = k0 + 16 + 8*(L/16) + (e-8)
__device__ __forceinline__ v16bf load_A_global(const float* base, int ld, int k0, int lane) {
  const float* p = base + (size_t)(lane & 15) * (size_t)ld + k0 + 8 * (lane >> 4);
  v16bf f;
#pragma unroll
  for (int g = 0; g < 2; ++g) {                    // two contiguous K-groups
    const float4 x0 = *(const float4*)(p + 16 * g);
    const float4 x1 = *(const float4*)(p + 16 * g + 4);
    f[8 * g + 0] = (__bf16)x0.x; f[8 * g + 1] = (__bf16)x0.y;
    f[8 * g + 2] = (__bf16)x0.z; f[8 * g + 3] = (__bf16)x0.w;
    f[8 * g + 4] = (__bf16)x1.x; f[8 * g + 5] = (__bf16)x1.y;
    f[8 * g + 6] = (__bf16)x1.z; f[8 * g + 7] = (__bf16)x1.w;
  }
  return f;
}

// A-fragment from a bf16 row-major LDS tile: both K-groups are 16 contiguous
// bytes -> exactly two ds_load_b128, zero converts.
__device__ __forceinline__ v16bf load_A_ldsb(const __bf16* st, int ld, int k0, int lane) {
  const __bf16* p = st + (lane & 15) * ld + k0 + 8 * (lane >> 4);
  const v8bf lo = *(const v8bf*)(p);
  const v8bf hi = *(const v8bf*)(p + 16);
  v16bf f;
#pragma unroll
  for (int i = 0; i < 8; ++i) { f[i] = lo[i]; f[8 + i] = hi[i]; }
  return f;
}

// ---- tile helpers ----------------------------------------------------------
template<int NT>
__device__ __forceinline__ void zero_acc(v8f* acc) {
#pragma unroll
  for (int t = 0; t < NT; ++t)
#pragma unroll
    for (int i = 0; i < 8; ++i) acc[t][i] = 0.f;
}

// C += A(16 x KC*32) x Wlds(KC*32 x NT*16).  kc-outer / t-inner: the NT WMMAs
// per k-chunk hit independent accumulators -> back-to-back XDL issue.
template<int NT, int KC>
__device__ __forceinline__ void gemm_lds(v8f* acc, const v16bf* A, const __bf16* wb, int lane) {
#pragma unroll
  for (int kc = 0; kc < KC; ++kc)
#pragma unroll
    for (int t = 0; t < NT; ++t) {
      const v16bf b = *(const v16bf*)(wb + (((t * KC + kc) * 32 + lane) << 4));
      acc[t] = __builtin_amdgcn_wmma_f32_16x16x32_bf16(
          false, A[kc], false, b, (short)0, acc[t], false, false);
    }
}

template<int NT, bool ACT>
__device__ __forceinline__ void bias_act(v8f* acc, const float* bias, int lane) {
  const int n16 = lane & 15;
#pragma unroll
  for (int t = 0; t < NT; ++t) {
    const float bb = bias[t * 16 + n16];
#pragma unroll
    for (int i = 0; i < 8; ++i) {
      const float x = acc[t][i] + bb;
      acc[t][i] = ACT ? leakyf(x) : x;
    }
  }
}

// C-tile layout -> row-major f32 LDS (M = i + 8*half, N = 16t + lane%16)
template<int NT>
__device__ __forceinline__ void store_C_lds(float* st, int ld, const v8f* acc, int lane) {
  const int n16 = lane & 15, half = lane >> 4;
#pragma unroll
  for (int t = 0; t < NT; ++t)
#pragma unroll
    for (int i = 0; i < 8; ++i)
      st[(i + 8 * half) * ld + t * 16 + n16] = acc[t][i];
}

// C-tile layout -> row-major bf16 LDS (convert once at store)
template<int NT>
__device__ __forceinline__ void store_C_ldsb(__bf16* st, int ld, const v8f* acc, int lane) {
  const int n16 = lane & 15, half = lane >> 4;
#pragma unroll
  for (int t = 0; t < NT; ++t)
#pragma unroll
    for (int i = 0; i < 8; ++i)
      st[(i + 8 * half) * ld + t * 16 + n16] = (__bf16)acc[t][i];
}

// e_k[n] = leaky([s_other_n | a_other_n] @ W_enc_sa[n] + b_enc_sa[n]), as A-frags.
// W_enc_sa[n] must already be staged (pre-swizzled bf16) in wb.
__device__ __forceinline__ void compute_ek(v16bf* eA, const float* s, const float* a,
                                           const __bf16* wb, const float* bsa,
                                           int n, int rowBase, int lane, __bf16* stb) {
  v16bf X[4];
  const float* sb = s + (size_t)rowBase * SROW + SD + (size_t)n * SD;  // 96 cols
  const float* ab = a + (size_t)rowBase * AROW + AD + (size_t)n * AD;  // 32 cols
  X[0] = load_A_global(sb, SROW, 0, lane);
  X[1] = load_A_global(sb, SROW, 32, lane);
  X[2] = load_A_global(sb, SROW, 64, lane);
  X[3] = load_A_global(ab, AROW, 0, lane);   // SA columns 96..127
  v8f acc[8]; zero_acc<8>(acc);
  gemm_lds<8, 4>(acc, X, wb, lane);
  bias_act<8, true>(acc, bsa + n * ATT, lane);
  store_C_ldsb<8>(stb, ATT, acc, lane);
  WAVE_LDS_FENCE();
#pragma unroll
  for (int kc = 0; kc < 4; ++kc) eA[kc] = load_A_ldsb(stb, ATT, kc * 32, lane);
}

// ---- fused forward kernel --------------------------------------------------
__global__ __launch_bounds__(NTHREADS, 1)
void attention_critic_fwd(
    const float* __restrict__ s,  const float* __restrict__ a,
    const float* __restrict__ W_enc_s,  const float* __restrict__ b_enc_s,
    const float* __restrict__ W_enc_sa, const float* __restrict__ b_enc_sa,
    const float* __restrict__ Wq, const float* __restrict__ Wk,
    const float* __restrict__ Wv, const float* __restrict__ bv,
    const float* __restrict__ Wo, const float* __restrict__ bo,
    const float* __restrict__ W_fc1, const float* __restrict__ b_fc1,
    const float* __restrict__ W_fc2, const float* __restrict__ b_fc2,
    float* __restrict__ out_q, float* __restrict__ out_allq)
{
  // block-shared pre-swizzled bf16 weight buffers
  __shared__ __align__(32) __bf16 WBUF0[ATT * ATT];   // per-agent W_enc_sa[n] (32 KB)
  __shared__ __align__(32) __bf16 WBUF1[ATT * ATT];   // stage-persistent weight (32 KB)
  // per-wave buffers
  __shared__ __align__(16) float  QLDS[WAVES_PER_BLK][16 * ATT];      // Q, f32
  __shared__ __align__(16) float  STG [WAVES_PER_BLK][16 * ATT];      // K / all_q, f32
  __shared__ __align__(32) __bf16 STGB[WAVES_PER_BLK][16 * ATT];      // A-frag shuffles
  __shared__ __align__(16) float  WL  [WAVES_PER_BLK][16 * 4 * NOTH]; // softmax weights

  const int lane = threadIdx.x & 31;
  const int wave = threadIdx.x >> 5;
  const int rowBase = blockIdx.x * ROWS_PER_BLK + wave * ROWS_PER_WAVE;
  float*  qlds = QLDS[wave];
  float*  st   = STG[wave];
  __bf16* stb  = STGB[wave];
  float*  wl   = WL[wave];
  const int n16 = lane & 15, half = lane >> 4;

  // ---- s_i = leaky(s[:, :96] @ W_enc_s + b_enc_s); keep bf16 A-frags ----
  v16bf siA[4];
  {
    stage_weight<SD, ATT>(WBUF1, W_enc_s, threadIdx.x);
    __syncthreads();
    v16bf X[3];
    const float* sb = s + (size_t)rowBase * SROW;
    X[0] = load_A_global(sb, SROW, 0, lane);
    X[1] = load_A_global(sb, SROW, 32, lane);
    X[2] = load_A_global(sb, SROW, 64, lane);
    v8f acc[8]; zero_acc<8>(acc);
    gemm_lds<8, 3>(acc, X, WBUF1, lane);
    bias_act<8, true>(acc, b_enc_s, lane);
    store_C_ldsb<8>(stb, ATT, acc, lane);
    WAVE_LDS_FENCE();
#pragma unroll
    for (int kc = 0; kc < 4; ++kc) siA[kc] = load_A_ldsb(stb, ATT, kc * 32, lane);
  }

  // ---- Q = s_i @ Wq -> per-wave LDS (row-major 16x128 f32) ----
  {
    STAGE(WBUF1, Wq, ATT, ATT);
    v8f acc[8]; zero_acc<8>(acc);
    gemm_lds<8, 4>(acc, siA, WBUF1, lane);
    store_C_lds<8>(qlds, ATT, acc, lane);
    WAVE_LDS_FENCE();
  }

  // ---- pass 1: raw scores per agent -> WL ----
  STAGE(WBUF1, Wk, ATT, ATT);            // Wk persists across the agent loop
#pragma unroll 1
  for (int n = 0; n < NOTH; ++n) {
    __syncthreads();
    stage_weight<ATT, ATT>(WBUF0, W_enc_sa + (size_t)n * ATT * ATT, threadIdx.x);
    if (n + 1 < NOTH) {                  // warm L2 for the next agent's weights
#pragma unroll
      for (int q = 0; q < 4; ++q)
        __builtin_prefetch(W_enc_sa + (size_t)(n + 1) * ATT * ATT +
                           (size_t)(threadIdx.x + q * NTHREADS) * 32, 0, 1);
    }
    __syncthreads();

    v16bf eA[4];
    compute_ek(eA, s, a, WBUF0, b_enc_sa, n, rowBase, lane, stb);
    v8f acc[8]; zero_acc<8>(acc);
    gemm_lds<8, 4>(acc, eA, WBUF1, lane);          // K_n
    store_C_lds<8>(st, ATT, acc, lane);
    WAVE_LDS_FENCE();
#pragma unroll
    for (int p = 0; p < 2; ++p) {                  // lane owns 2 (row,head) pairs
      const int row = n16, head = half + 2 * p;
      const float4* qr = (const float4*)(qlds + row * ATT + head * 32);
      const float4* kr = (const float4*)(st   + row * ATT + head * 32);
      float d = 0.f;
#pragma unroll
      for (int j = 0; j < 8; ++j) {
        const float4 qv = qr[j], kv = kr[j];
        d += qv.x * kv.x + qv.y * kv.y + qv.z * kv.z + qv.w * kv.w;
      }
      wl[(row * 4 + head) * NOTH + n] = d * 0.17677669529663687f; // 1/sqrt(32)
    }
    WAVE_LDS_FENCE();
  }

  // ---- softmax over the 7 agents, in place in WL (per-wave) ----
#pragma unroll
  for (int p = 0; p < 2; ++p) {
    const int row = n16, head = half + 2 * p;
    float* w = wl + (row * 4 + head) * NOTH;
    float m = w[0];
#pragma unroll
    for (int n = 1; n < NOTH; ++n) m = fmaxf(m, w[n]);
    float sum = 0.f;
#pragma unroll
    for (int n = 0; n < NOTH; ++n) { const float e = __expf(w[n] - m); w[n] = e; sum += e; }
    const float inv = 1.f / sum;
#pragma unroll
    for (int n = 0; n < NOTH; ++n) w[n] *= inv;
  }
  WAVE_LDS_FENCE();

  // ---- pass 2: attn = sum_n w_n * leaky(e_k @ Wv + bv), in C-tile layout ----
  v8f attn[8]; zero_acc<8>(attn);
  STAGE(WBUF1, Wv, ATT, ATT);            // Wv persists across the agent loop
#pragma unroll 1
  for (int n = 0; n < NOTH; ++n) {
    __syncthreads();
    stage_weight<ATT, ATT>(WBUF0, W_enc_sa + (size_t)n * ATT * ATT, threadIdx.x);
    __syncthreads();

    v16bf eA[4];
    compute_ek(eA, s, a, WBUF0, b_enc_sa, n, rowBase, lane, stb);
    v8f vac[8]; zero_acc<8>(vac);
    gemm_lds<8, 4>(vac, eA, WBUF1, lane);          // V_n
    bias_act<8, true>(vac, bv, lane);
#pragma unroll
    for (int t = 0; t < 8; ++t) {
      const int head = t >> 1;                     // cols 16t..16t+15 -> head t/2
#pragma unroll
      for (int i = 0; i < 8; ++i) {
        const int M = i + 8 * half;
        attn[t][i] += wl[(M * 4 + head) * NOTH + n] * vac[t][i];
      }
    }
  }

  // ---- other = attn @ Wo + bo ----
  v16bf oA[4];
  {
    STAGE(WBUF1, Wo, ATT, ATT);
    store_C_ldsb<8>(stb, ATT, attn, lane);
    WAVE_LDS_FENCE();
    v16bf aA[4];
#pragma unroll
    for (int kc = 0; kc < 4; ++kc) aA[kc] = load_A_ldsb(stb, ATT, kc * 32, lane);
    v8f acc[8]; zero_acc<8>(acc);
    gemm_lds<8, 4>(acc, aA, WBUF1, lane);
    bias_act<8, false>(acc, bo, lane);
    store_C_ldsb<8>(stb, ATT, acc, lane);
    WAVE_LDS_FENCE();
#pragma unroll
    for (int kc = 0; kc < 4; ++kc) oA[kc] = load_A_ldsb(stb, ATT, kc * 32, lane);
  }

  // ---- h = leaky([s_i | other] @ W_fc1 + b_fc1) (two 128-row halves) ----
  v16bf hA[4];
  {
    v8f acc[8]; zero_acc<8>(acc);
    STAGE(WBUF1, W_fc1, ATT, ATT);                       // X rows 0..127
    gemm_lds<8, 4>(acc, siA, WBUF1, lane);
    STAGE(WBUF1, W_fc1 + ATT * ATT, ATT, ATT);           // X rows 128..255
    gemm_lds<8, 4>(acc, oA, WBUF1, lane);
    bias_act<8, true>(acc, b_fc1, lane);
    store_C_ldsb<8>(stb, ATT, acc, lane);
    WAVE_LDS_FENCE();
#pragma unroll
    for (int kc = 0; kc < 4; ++kc) hA[kc] = load_A_ldsb(stb, ATT, kc * 32, lane);
  }

  // ---- all_q = h @ W_fc2 + b_fc2 (128 -> 32), write out + stash in LDS ----
  {
    STAGE(WBUF1, W_fc2, ATT, AD);
    v8f acc2[2]; zero_acc<2>(acc2);
    gemm_lds<2, 4>(acc2, hA, WBUF1, lane);
    bias_act<2, false>(acc2, b_fc2, lane);
#pragma unroll
    for (int t = 0; t < 2; ++t)
#pragma unroll
      for (int i = 0; i < 8; ++i) {
        const int M = i + 8 * half;
        const int col = t * 16 + n16;
        const int gr = rowBase + M;
        out_allq[(size_t)gr * 32 + col] = acc2[t][i];
        st[M * 32 + col] = acc2[t][i];
      }
    WAVE_LDS_FENCE();
  }

  // ---- q = all_q[argmax(a[:, :32])] (first-max, as jnp.argmax) ----
  if (lane < 16) {
    const int gr = rowBase + lane;
    const float* ar = a + (size_t)gr * AROW;
    float best = ar[0]; int bj = 0;
#pragma unroll 1
    for (int j = 1; j < 32; ++j) { const float v = ar[j]; if (v > best) { best = v; bj = j; } }
    out_q[gr] = st[lane * 32 + bj];
  }
}

extern "C" void kernel_launch(void* const* d_in, const int* in_sizes, int n_in,
                              void* d_out, int out_size, void* d_ws, size_t ws_size,
                              hipStream_t stream) {
  const float* s        = (const float*)d_in[0];
  const float* a        = (const float*)d_in[1];
  const float* W_enc_s  = (const float*)d_in[2];
  const float* b_enc_s  = (const float*)d_in[3];
  const float* W_enc_sa = (const float*)d_in[4];
  const float* b_enc_sa = (const float*)d_in[5];
  const float* Wq       = (const float*)d_in[6];
  const float* Wk       = (const float*)d_in[7];
  const float* Wv       = (const float*)d_in[8];
  const float* bv       = (const float*)d_in[9];
  const float* Wo       = (const float*)d_in[10];
  const float* bo       = (const float*)d_in[11];
  const float* W_fc1    = (const float*)d_in[12];
  const float* b_fc1    = (const float*)d_in[13];
  const float* W_fc2    = (const float*)d_in[14];
  const float* b_fc2    = (const float*)d_in[15];

  const int Btot = in_sizes[0] / SROW;            // 32768
  float* out_q    = (float*)d_out;                // (B,1) first
  float* out_allq = (float*)d_out + Btot;         // then (B,32)

  const int blocks = Btot / ROWS_PER_BLK;         // 512
  attention_critic_fwd<<<blocks, NTHREADS, 0, stream>>>(
      s, a, W_enc_s, b_enc_s, W_enc_sa, b_enc_sa,
      Wq, Wk, Wv, bv, Wo, bo, W_fc1, b_fc1, W_fc2, b_fc2,
      out_q, out_allq);
}
